// BitLinear_33174327394884
// MI455X (gfx1250) — matrix operand order repaired and checked by reading, not compiled, not run on hardware
//
#include <hip/hip_runtime.h>

typedef __attribute__((ext_vector_type(16))) _Float16 v16h;
typedef __attribute__((ext_vector_type(8)))  _Float16 v8h;
typedef __attribute__((ext_vector_type(4)))  _Float16 v4h;
typedef __attribute__((ext_vector_type(8)))  float    v8f;

// Problem constants (from reference setup_inputs)
constexpr int Mdim = 4 * 4096;   // B*S rows
constexpr int Ndim = 2048;       // D_OUT
constexpr int Kdim = 2048;       // D_IN

static __device__ __forceinline__ float waveSum(float v) {
#pragma unroll
    for (int o = 16; o > 0; o >>= 1) v += __shfl_down(v, o, 32);
    return v;
}

static __device__ __forceinline__ v16h cat8(v8h lo, v8h hi) {
    v16h r;
#pragma unroll
    for (int i = 0; i < 8; ++i) { r[i] = lo[i]; r[i + 8] = hi[i]; }
    return r;
}

// ---- Pass 1: partial sums of |W| ------------------------------------------
__global__ void __launch_bounds__(256)
k_absmean_partial(const float* __restrict__ w, float* __restrict__ part, int n) {
    float s = 0.f;
    for (int i = blockIdx.x * blockDim.x + threadIdx.x; i < n; i += gridDim.x * blockDim.x)
        s += fabsf(w[i]);
    __shared__ float sm[8];
    float wsum = waveSum(s);
    int wv = threadIdx.x >> 5, ln = threadIdx.x & 31;
    if (ln == 0) sm[wv] = wsum;
    __syncthreads();
    if (wv == 0) {
        float a = (ln < 8) ? sm[ln] : 0.f;
        a = waveSum(a);
        if (ln == 0) part[blockIdx.x] = a;
    }
}

// ---- Pass 2: finalize scale = max(mean|W|, 1e-8) --------------------------
__global__ void __launch_bounds__(256)
k_absmean_final(const float* __restrict__ part, float* __restrict__ scale) {
    float s = part[threadIdx.x] + part[threadIdx.x + 256];
    __shared__ float sm[8];
    float wsum = waveSum(s);
    int wv = threadIdx.x >> 5, ln = threadIdx.x & 31;
    if (ln == 0) sm[wv] = wsum;
    __syncthreads();
    if (threadIdx.x == 0) {
        float t = 0.f;
#pragma unroll
        for (int i = 0; i < 8; ++i) t += sm[i];
        float m = t * (1.0f / (float)((long long)Ndim * Kdim));
        scale[0] = fmaxf(m, 1e-8f);
    }
}

// ---- Pass 3: ternary-quantize W to f16 ------------------------------------
__global__ void __launch_bounds__(256)
k_quantize(const float* __restrict__ w, const float* __restrict__ scale,
           _Float16* __restrict__ qw, int n4) {
    int i = blockIdx.x * blockDim.x + threadIdx.x;
    if (i >= n4) return;
    const float inv = 1.0f / scale[0];
    const float4 v = ((const float4*)w)[i];
    v4h q;
    q[0] = (_Float16)fminf(fmaxf(rintf(v.x * inv), -1.f), 1.f);
    q[1] = (_Float16)fminf(fmaxf(rintf(v.y * inv), -1.f), 1.f);
    q[2] = (_Float16)fminf(fmaxf(rintf(v.z * inv), -1.f), 1.f);
    q[3] = (_Float16)fminf(fmaxf(rintf(v.w * inv), -1.f), 1.f);
    ((v4h*)qw)[i] = q;
}

// ---- Pass 4: non-affine LayerNorm over K, write f16 -----------------------
__global__ void __launch_bounds__(256)
k_layernorm(const float* __restrict__ x, _Float16* __restrict__ x16) {
    const int row = blockIdx.x;
    const float* xr = x + (size_t)row * Kdim;
    const int base = threadIdx.x * 8;           // 256 threads * 8 = 2048
    const float4 a = ((const float4*)(xr + base))[0];
    const float4 b = ((const float4*)(xr + base + 4))[0];
    float s  = a.x + a.y + a.z + a.w + b.x + b.y + b.z + b.w;
    float s2 = a.x*a.x + a.y*a.y + a.z*a.z + a.w*a.w
             + b.x*b.x + b.y*b.y + b.z*b.z + b.w*b.w;
    __shared__ float smS[8], smQ[8];
    float rs = waveSum(s), rq = waveSum(s2);
    int wv = threadIdx.x >> 5, ln = threadIdx.x & 31;
    if (ln == 0) { smS[wv] = rs; smQ[wv] = rq; }
    __syncthreads();
    if (threadIdx.x == 0) {
        float ts = 0.f, tq = 0.f;
#pragma unroll
        for (int i = 0; i < 8; ++i) { ts += smS[i]; tq += smQ[i]; }
        smS[0] = ts; smQ[0] = tq;
    }
    __syncthreads();
    const float mean = smS[0] * (1.0f / Kdim);
    const float var  = smQ[0] * (1.0f / Kdim) - mean * mean;
    const float r = rsqrtf(var + 1e-5f);
    v8h o;
    o[0] = (_Float16)((a.x - mean) * r); o[1] = (_Float16)((a.y - mean) * r);
    o[2] = (_Float16)((a.z - mean) * r); o[3] = (_Float16)((a.w - mean) * r);
    o[4] = (_Float16)((b.x - mean) * r); o[5] = (_Float16)((b.y - mean) * r);
    o[6] = (_Float16)((b.z - mean) * r); o[7] = (_Float16)((b.w - mean) * r);
    *(v8h*)(x16 + (size_t)row * Kdim + base) = o;
}

// ---- Pass 5: WMMA GEMM: out[m,n] = sum_k A[m,k]*Bq[n,k] -------------------
// Block: 256 threads (8 wave32) -> 128x128 output tile.
// Wave (wr 0..3, wc 0..1) -> 32x64 region = 2x4 accumulators of 16x16.
__global__ void __launch_bounds__(256)
k_gemm(const _Float16* __restrict__ A, const _Float16* __restrict__ Bq,
       float* __restrict__ C) {
    const int lane = threadIdx.x & 31;
    const int wave = threadIdx.x >> 5;
    const int wr = wave >> 1;
    const int wc = wave & 1;
    const int mBase = blockIdx.y * 128 + wr * 32;
    const int nBase = blockIdx.x * 128 + wc * 64;
    const int l15 = lane & 15;
    const int hi  = lane >> 4;

    v8f acc[2][4] = {};

    // A: lanes 0-15 -> row M=l15, halfs = K{0..7,16..23}; high lanes +8
    const _Float16* aRow0 = A + (size_t)(mBase + l15) * Kdim + (hi ? 8 : 0);
    const _Float16* aRow1 = aRow0 + (size_t)16 * Kdim;
    // B: lanes 0-15 -> col N=l15, halfs = K{0..7,16..23}; high lanes +8
    const _Float16* bCol[4];
#pragma unroll
    for (int j = 0; j < 4; ++j)
        bCol[j] = Bq + (size_t)(nBase + j * 16 + l15) * Kdim + (hi ? 8 : 0);

    for (int k0 = 0; k0 < Kdim; k0 += 32) {
        v16h af[2], bf[4];
        {
            v8h lo = *(const v8h*)(aRow0 + k0);
            v8h h2 = *(const v8h*)(aRow0 + k0 + 16);
            af[0] = cat8(lo, h2);
            lo = *(const v8h*)(aRow1 + k0);
            h2 = *(const v8h*)(aRow1 + k0 + 16);
            af[1] = cat8(lo, h2);
        }
#pragma unroll
        for (int j = 0; j < 4; ++j) {
            v8h lo = *(const v8h*)(bCol[j] + k0);
            v8h h2 = *(const v8h*)(bCol[j] + k0 + 16);
            bf[j] = cat8(lo, h2);
        }
#pragma unroll
        for (int i = 0; i < 2; ++i)
#pragma unroll
            for (int j = 0; j < 4; ++j)
                acc[i][j] = __builtin_amdgcn_wmma_f32_16x16x32_f16(
                    false, af[i], false, bf[j], (short)0, acc[i][j], false, false);

        if (k0 + 64 < Kdim) {
            __builtin_prefetch(aRow0 + k0 + 64, 0, 3);   // global_prefetch_b8
            __builtin_prefetch(bCol[0] + k0 + 64, 0, 3);
        }
    }

    // C/D layout: VGPR d, lanes 0-15 -> (M=d, N=lane); lanes 16-31 -> (M=d+8)
#pragma unroll
    for (int i = 0; i < 2; ++i) {
        const int rBase = mBase + i * 16 + hi * 8;
#pragma unroll
        for (int j = 0; j < 4; ++j) {
            const int col = nBase + j * 16 + l15;
            float* outp = C + (size_t)rBase * Ndim + col;
#pragma unroll
            for (int d = 0; d < 8; ++d)
                outp[(size_t)d * Ndim] = acc[i][j][d];
        }
    }
}

extern "C" void kernel_launch(void* const* d_in, const int* in_sizes, int n_in,
                              void* d_out, int out_size, void* d_ws, size_t ws_size,
                              hipStream_t stream) {
    (void)in_sizes; (void)n_in; (void)out_size; (void)ws_size;
    const float* x = (const float*)d_in[0];   // [4,4096,2048] f32
    const float* w = (const float*)d_in[1];   // [2048,2048]  f32
    float* out = (float*)d_out;               // [4,4096,2048] f32

    char* ws = (char*)d_ws;
    float*    scale    = (float*)ws;                                  // 4 B
    float*    partials = (float*)(ws + 256);                          // 512 f32
    _Float16* x16      = (_Float16*)(ws + 4096);                      // 64 MiB
    _Float16* qw       = (_Float16*)(ws + 4096 + (size_t)Mdim * Kdim * 2); // 8 MiB

    const int nW = Ndim * Kdim;
    k_absmean_partial<<<512, 256, 0, stream>>>(w, partials, nW);
    k_absmean_final  <<<1,   256, 0, stream>>>(partials, scale);
    k_quantize       <<<(nW / 4 + 255) / 256, 256, 0, stream>>>(w, scale, qw, nW / 4);
    k_layernorm      <<<Mdim, 256, 0, stream>>>(x, x16);

    dim3 grid(Ndim / 128, Mdim / 128);   // (16, 128)
    k_gemm<<<grid, 256, 0, stream>>>(x16, qw, out);
}
